// AttnNCRFDecoder_1537598292340
// MI455X (gfx1250) — matrix operand order repaired and checked by reading, not compiled, not run on hardware
//
#include <hip/hip_runtime.h>
#include <hip/hip_fp16.h>
#include <math.h>

#define BB 64
#define SS 512
#define DD 768
#define HH 3
#define DKK 64
#define TAG 33
#define NBEST 8
#define START 31
#define STOP 32

typedef __attribute__((ext_vector_type(16))) _Float16 v16h;
typedef __attribute__((ext_vector_type(8)))  float    v8f;

__device__ __forceinline__ v8f wmma_f16(v16h a, v16h b, v8f c) {
  // D = A(16x32 f16) * B(32x16 f16) + C(16x16 f32)
  return __builtin_amdgcn_wmma_f32_16x16x32_f16(false, a, false, b, (short)0, c, false, false);
}

__device__ __forceinline__ v8f zero8() {
  v8f z;
#pragma unroll
  for (int i = 0; i < 8; ++i) z[i] = 0.0f;
  return z;
}

// CDNA5 async data mover: copy 16B per lane from global directly into LDS.
// Generic pointers to __shared__ carry the LDS byte offset in their low 32 bits
// (ISA 10.2 aperture mapping), which is exactly the VDST operand the async op wants.
__device__ __forceinline__ void async_b128(void* lds_dst, const void* gsrc) {
  unsigned ldso = (unsigned)(unsigned long long)lds_dst;
  asm volatile("global_load_async_to_lds_b128 %0, %1, off"
               :: "v"(ldso), "v"(gsrc) : "memory");
}
__device__ __forceinline__ void wait_async0() {
  asm volatile("s_wait_asynccnt 0x0" ::: "memory");
}

// A-fragment (16x32, f16) from LDS tile stored row-major [m][k], row stride ld (halfs).
__device__ __forceinline__ v16h frag_a(const _Float16* p, int ld, int lane) {
  const int m = lane & 15;
  const int kb = (lane & 16) ? 16 : 0;
  const _Float16* row = p + m * ld + kb;
  v16h a;
#pragma unroll
  for (int e = 0; e < 16; ++e) a[e] = row[e];
  return a;
}

// B-fragment (32x16, f16) from LDS tile stored transposed [n][k], row stride ld (halfs).
__device__ __forceinline__ v16h frag_b(const _Float16* p, int ld, int lane) {
  const int n = lane & 15;
  const int kb = (lane & 16) ? 16 : 0;
  const _Float16* row = p + n * ld + kb;
  v16h b;
#pragma unroll
  for (int e = 0; e < 16; ++e) b[e] = row[e];
  return b;
}

// ---------------- QKV projection: per (b,h): x[b] (512x768) x w (768x64) ----------------
__global__ __launch_bounds__(256) void k_qkv(
    const float* __restrict__ x,
    const float* __restrict__ wq, const float* __restrict__ wk, const float* __restrict__ wv,
    _Float16* __restrict__ qo, _Float16* __restrict__ ko, _Float16* __restrict__ vTo)
{
  __shared__ _Float16 lA[128][40];
  __shared__ _Float16 lB[3][64][40];
  const int bh = blockIdx.x, b = bh / HH, h = bh % HH;
  const int rowBlk = blockIdx.y * 128;
  const int tid = threadIdx.x, lane = tid & 31, w = tid >> 5;
  const float* W[3] = { wq + (size_t)h * DD * DKK, wk + (size_t)h * DD * DKK,
                        wv + (size_t)h * DD * DKK };
  v8f acc[3][4];
#pragma unroll
  for (int m = 0; m < 3; ++m)
#pragma unroll
    for (int n = 0; n < 4; ++n) acc[m][n] = zero8();

  for (int kb = 0; kb < DD; kb += 32) {
    for (int i = tid; i < 128 * 32; i += 256) {
      int r = i >> 5, c = i & 31;
      lA[r][c] = (_Float16)x[((size_t)(b * SS + rowBlk + r)) * DD + kb + c];
    }
    if (kb + 32 < DD)
      __builtin_prefetch(&x[((size_t)(b * SS + rowBlk + (tid >> 1))) * DD + kb + 32], 0, 3);
    for (int i = tid; i < 3 * 64 * 32; i += 256) {
      int m = i >> 11, rem = i & 2047, kk = rem >> 6, n = rem & 63;
      lB[m][n][kk] = (_Float16)W[m][(size_t)(kb + kk) * DKK + n];
    }
    __syncthreads();
    v16h a = frag_a(&lA[w * 16][0], 40, lane);
#pragma unroll
    for (int m = 0; m < 3; ++m)
#pragma unroll
      for (int n = 0; n < 4; ++n) {
        v16h bf = frag_b(&lB[m][n * 16][0], 40, lane);
        acc[m][n] = wmma_f16(a, bf, acc[m][n]);
      }
    __syncthreads();
  }
  const int roff = (lane & 16) ? 8 : 0, c0 = lane & 15;
#pragma unroll
  for (int n = 0; n < 4; ++n)
#pragma unroll
    for (int r = 0; r < 8; ++r) {
      int s = rowBlk + w * 16 + r + roff;
      int cc = n * 16 + c0;
      qo[((size_t)bh * SS + s) * DKK + cc] = (_Float16)(acc[0][n][r] * 0.125f); // 1/sqrt(64)
      ko[((size_t)bh * SS + s) * DKK + cc] = (_Float16)acc[1][n][r];
      vTo[((size_t)bh * DKK + cc) * SS + s] = (_Float16)acc[2][n][r];
    }
}

// ---------------- Flash attention: block = (b,h, q-tile of 128), K & V^T in LDS ----------------
__global__ __launch_bounds__(256) void k_attn(
    const _Float16* __restrict__ q, const _Float16* __restrict__ k,
    const _Float16* __restrict__ vT, _Float16* __restrict__ attn)
{
  __shared__ _Float16 lK[512][72];   // [key][dk]  (B-frag layout for Q*K^T)
  __shared__ _Float16 lVT[64][520];  // [dv][key]  (B-frag layout for P*V)
  __shared__ _Float16 lQ[128][72];
  __shared__ _Float16 lP[8][16][40]; // per-wave P scratch (A-layout round trip)
  const int bh = blockIdx.x, b = bh / HH, h = bh % HH;
  const int qt = blockIdx.y;
  const int tid = threadIdx.x, lane = tid & 31, w = tid >> 5;

  for (int i = tid; i < 512 * 8; i += 256) {
    int r = i >> 3, c8 = (i & 7) * 8;
    async_b128(&lK[r][c8], &k[((size_t)bh * SS + r) * DKK + c8]);
  }
  for (int i = tid; i < 64 * 64; i += 256) {
    int r = i >> 6, c8 = (i & 63) * 8;
    async_b128(&lVT[r][c8], &vT[((size_t)bh * DKK + r) * SS + c8]);
  }
  for (int i = tid; i < 128 * 8; i += 256) {
    int r = i >> 3, c8 = (i & 7) * 8;
    async_b128(&lQ[r][c8], &q[((size_t)bh * SS + qt * 128 + r) * DKK + c8]);
  }
  wait_async0();
  __syncthreads();

  v16h a0 = frag_a(&lQ[w * 16][0], 72, lane);
  v16h a1 = frag_a(&lQ[w * 16][32], 72, lane);
  float mi[8], li[8];
  v8f o[4];
#pragma unroll
  for (int r = 0; r < 8; ++r) { mi[r] = -3.0e38f; li[r] = 0.0f; }
#pragma unroll
  for (int n = 0; n < 4; ++n) o[n] = zero8();
  const int roff = (lane & 16) ? 8 : 0, c0 = lane & 15;

  for (int jt = 0; jt < 16; ++jt) {
    const int j0 = jt * 32;
    v8f s0 = zero8(), s1 = zero8();
    s0 = wmma_f16(a0, frag_b(&lK[j0][0], 72, lane), s0);
    s0 = wmma_f16(a1, frag_b(&lK[j0][32], 72, lane), s0);
    s1 = wmma_f16(a0, frag_b(&lK[j0 + 16][0], 72, lane), s1);
    s1 = wmma_f16(a1, frag_b(&lK[j0 + 16][32], 72, lane), s1);
#pragma unroll
    for (int r = 0; r < 8; ++r) {
      float v = fmaxf(s0[r], s1[r]);
#pragma unroll
      for (int d = 1; d < 16; d <<= 1) v = fmaxf(v, __shfl_xor(v, d, 32));
      float mn = fmaxf(mi[r], v);
      float corr = __expf(mi[r] - mn);
      float p0 = __expf(s0[r] - mn);
      float p1 = __expf(s1[r] - mn);
      float rs = p0 + p1;
#pragma unroll
      for (int d = 1; d < 16; d <<= 1) rs += __shfl_xor(rs, d, 32);
      li[r] = li[r] * corr + rs;
      mi[r] = mn;
#pragma unroll
      for (int n = 0; n < 4; ++n) o[n][r] = o[n][r] * corr;
      lP[w][r + roff][c0] = (_Float16)p0;
      lP[w][r + roff][16 + c0] = (_Float16)p1;
    }
    v16h ap = frag_a(&lP[w][0][0], 40, lane); // same-wave LDS: in-order, no barrier
#pragma unroll
    for (int n = 0; n < 4; ++n)
      o[n] = wmma_f16(ap, frag_b(&lVT[n * 16][j0], 520, lane), o[n]);
  }
#pragma unroll
  for (int n = 0; n < 4; ++n)
#pragma unroll
    for (int r = 0; r < 8; ++r) {
      int s = qt * 128 + w * 16 + r + roff;
      attn[((size_t)(b * SS + s)) * (HH * DKK) + h * DKK + n * 16 + c0] =
          (_Float16)(o[n][r] / li[r]);
    }
}

// ---------------- proj (32768x192 @ 192x768) + bias + residual ----------------
__global__ __launch_bounds__(256) void k_proj(
    const _Float16* __restrict__ attn, const float* __restrict__ pw,
    const float* __restrict__ pb, const float* __restrict__ x, float* __restrict__ post)
{
  __shared__ _Float16 lA[128][200];
  __shared__ _Float16 lB[64][200];
  const int rows = blockIdx.x * 128, nBase = blockIdx.y * 64;
  const int tid = threadIdx.x, lane = tid & 31, w = tid >> 5;
  for (int i = tid; i < 128 * 24; i += 256) {
    int r = i / 24, c8 = (i % 24) * 8;
    async_b128(&lA[r][c8], &attn[((size_t)(rows + r)) * 192 + c8]);
  }
  for (int i = tid; i < 64 * 192; i += 256) {
    int kk = i >> 6, n = i & 63;
    lB[n][kk] = (_Float16)pw[(size_t)kk * DD + nBase + n];
  }
  wait_async0();
  __syncthreads();
  v8f acc[4];
#pragma unroll
  for (int n = 0; n < 4; ++n) acc[n] = zero8();
  for (int kc = 0; kc < 6; ++kc) {
    v16h a = frag_a(&lA[w * 16][kc * 32], 200, lane);
#pragma unroll
    for (int n = 0; n < 4; ++n)
      acc[n] = wmma_f16(a, frag_b(&lB[n * 16][kc * 32], 200, lane), acc[n]);
  }
  const int roff = (lane & 16) ? 8 : 0, c0 = lane & 15;
#pragma unroll
  for (int n = 0; n < 4; ++n)
#pragma unroll
    for (int r = 0; r < 8; ++r) {
      int row = rows + w * 16 + r + roff, cc = nBase + n * 16 + c0;
      post[(size_t)row * DD + cc] = acc[n][r] + pb[cc] + x[(size_t)row * DD + cc];
    }
}

// ---------------- LayerNorm (ddof=1, (sigma+eps) denom), one wave per row ----------------
__global__ __launch_bounds__(256) void k_ln(
    const float* __restrict__ post, const float* __restrict__ g,
    const float* __restrict__ bb, _Float16* __restrict__ y16)
{
  const int tid = threadIdx.x, lane = tid & 31, w = tid >> 5;
  const int row = blockIdx.x * 8 + w;
  const float* pr = post + (size_t)row * DD;
  float v[24];
  float sum = 0.0f;
#pragma unroll
  for (int i = 0; i < 24; ++i) { v[i] = pr[lane + i * 32]; sum += v[i]; }
#pragma unroll
  for (int d = 1; d < 32; d <<= 1) sum += __shfl_xor(sum, d, 32);
  float mu = sum * (1.0f / 768.0f);
  float ss = 0.0f;
#pragma unroll
  for (int i = 0; i < 24; ++i) { float t = v[i] - mu; ss += t * t; }
#pragma unroll
  for (int d = 1; d < 32; d <<= 1) ss += __shfl_xor(ss, d, 32);
  float inv = 1.0f / (sqrtf(ss * (1.0f / 767.0f)) + 1e-3f);
#pragma unroll
  for (int i = 0; i < 24; ++i) {
    int c = lane + i * 32;
    y16[(size_t)row * DD + c] = (_Float16)((v[i] - mu) * inv * g[c] + bb[c]);
  }
}

// ---------------- lin1 (768->384) + tanh ----------------
__global__ __launch_bounds__(256) void k_lin1(
    const _Float16* __restrict__ y16, const float* __restrict__ w1,
    const float* __restrict__ b1, _Float16* __restrict__ h1)
{
  __shared__ _Float16 lA[128][40];
  __shared__ _Float16 lB[64][40];
  const int rows = blockIdx.x * 128, nBase = blockIdx.y * 64;
  const int tid = threadIdx.x, lane = tid & 31, w = tid >> 5;
  v8f acc[4];
#pragma unroll
  for (int n = 0; n < 4; ++n) acc[n] = zero8();
  for (int kb = 0; kb < DD; kb += 32) {
    for (int i = tid; i < 128 * 4; i += 256) {
      int r = i >> 2, c8 = (i & 3) * 8;
      async_b128(&lA[r][c8], &y16[((size_t)(rows + r)) * DD + kb + c8]);
    }
    for (int i = tid; i < 64 * 32; i += 256) {
      int kk = i >> 6, n = i & 63;
      lB[n][kk] = (_Float16)w1[(size_t)(kb + kk) * 384 + nBase + n];
    }
    wait_async0();
    __syncthreads();
    v16h a = frag_a(&lA[w * 16][0], 40, lane);
#pragma unroll
    for (int n = 0; n < 4; ++n)
      acc[n] = wmma_f16(a, frag_b(&lB[n * 16][0], 40, lane), acc[n]);
    __syncthreads();
  }
  const int roff = (lane & 16) ? 8 : 0, c0 = lane & 15;
#pragma unroll
  for (int n = 0; n < 4; ++n)
#pragma unroll
    for (int r = 0; r < 8; ++r) {
      int row = rows + w * 16 + r + roff, cc = nBase + n * 16 + c0;
      h1[(size_t)row * 384 + cc] = (_Float16)tanhf(acc[n][r] + b1[cc]);
    }
}

// ---------------- lin2 (384->33), N padded to 48 ----------------
__global__ __launch_bounds__(256) void k_lin2(
    const _Float16* __restrict__ h1, const float* __restrict__ w2,
    const float* __restrict__ b2, float* __restrict__ logits)
{
  __shared__ _Float16 lA[128][40];
  __shared__ _Float16 lB[48][40];
  const int rows = blockIdx.x * 128;
  const int tid = threadIdx.x, lane = tid & 31, w = tid >> 5;
  v8f acc[3];
#pragma unroll
  for (int n = 0; n < 3; ++n) acc[n] = zero8();
  for (int kb = 0; kb < 384; kb += 32) {
    for (int i = tid; i < 128 * 4; i += 256) {
      int r = i >> 2, c8 = (i & 3) * 8;
      async_b128(&lA[r][c8], &h1[((size_t)(rows + r)) * 384 + kb + c8]);
    }
    for (int i = tid; i < 48 * 32; i += 256) {
      int kk = i / 48, n = i % 48;
      lB[n][kk] = (n < TAG) ? (_Float16)w2[(size_t)(kb + kk) * TAG + n] : (_Float16)0.0f;
    }
    wait_async0();
    __syncthreads();
    v16h a = frag_a(&lA[w * 16][0], 40, lane);
#pragma unroll
    for (int n = 0; n < 3; ++n)
      acc[n] = wmma_f16(a, frag_b(&lB[n * 16][0], 40, lane), acc[n]);
    __syncthreads();
  }
  const int roff = (lane & 16) ? 8 : 0, c0 = lane & 15;
#pragma unroll
  for (int n = 0; n < 3; ++n)
#pragma unroll
    for (int r = 0; r < 8; ++r) {
      int row = rows + w * 16 + r + roff, cc = n * 16 + c0;
      if (cc < TAG) logits[(size_t)row * TAG + cc] = acc[n][r] + b2[cc];
    }
}

// ---------------- Viterbi NBEST DP + backtrace, one block per batch ----------------
__global__ __launch_bounds__(64) void k_viterbi(
    const float* __restrict__ logits, const int* __restrict__ mask,
    const float* __restrict__ trans, short* __restrict__ bps, float* __restrict__ out)
{
  __shared__ float trs[TAG][TAG];
  __shared__ float part[TAG][NBEST];
  __shared__ float npart[TAG][NBEST];
  __shared__ float lpart[TAG][NBEST];
  __shared__ int sh_len;
  const int b = blockIdx.x, tid = threadIdx.x;
  for (int i = tid; i < TAG * TAG; i += 64) trs[i / TAG][i % TAG] = trans[i];
  if (tid == 0) {
    int len = 0;
    for (int t = 0; t < SS; ++t) len += mask[b * SS + t];
    sh_len = len;
  }
  __syncthreads();
  const int last_pos = sh_len - 1;
  if (tid < TAG) {
    part[tid][0] = logits[(size_t)b * SS * TAG + tid] + trs[START][tid];
#pragma unroll
    for (int n = 1; n < NBEST; ++n) part[tid][n] = -1.0e30f;
  }
  __syncthreads();

  for (int t = 1; t < SS; ++t) {
    const int m_t = mask[b * SS + t];
    if (tid < TAG) {
      const int j = tid;
      float lg = logits[((size_t)b * SS + t) * TAG + j];
      float vv[NBEST]; int ii[NBEST];
#pragma unroll
      for (int n = 0; n < NBEST; ++n) { vv[n] = -3.3e38f; ii[n] = 0; }
      for (int i = 0; i < TAG; ++i) {
        float tr = trs[i][j];
        for (int n = 0; n < NBEST; ++n) {     // part rows sorted desc -> prune
          float cv = tr + part[i][n];
          if (cv <= vv[NBEST - 1]) break;     // ties keep lower flat index
          int kp = NBEST - 1;
          while (kp > 0 && vv[kp - 1] < cv) {
            vv[kp] = vv[kp - 1]; ii[kp] = ii[kp - 1]; --kp;
          }
          vv[kp] = cv; ii[kp] = i * NBEST + n;
        }
      }
      size_t bo = (((size_t)(t - 1) * BB + b) * TAG + j) * NBEST;
#pragma unroll
      for (int n = 0; n < NBEST; ++n) {
        npart[j][n] = vv[n] + lg;
        bps[bo + n] = (short)(m_t ? ii[n] : 0);
      }
    }
    __syncthreads();
    if (tid < TAG) {
#pragma unroll
      for (int n = 0; n < NBEST; ++n) part[tid][n] = npart[tid][n];
      if (t == last_pos)
#pragma unroll
        for (int n = 0; n < NBEST; ++n) lpart[tid][n] = npart[tid][n];
    }
    __syncthreads();
  }

  if (tid == 0) {
    float vv[NBEST]; int ii[NBEST];
#pragma unroll
    for (int n = 0; n < NBEST; ++n) { vv[n] = -3.3e38f; ii[n] = 0; }
    for (int i = 0; i < TAG; ++i) {
      float tr = trs[i][STOP];
      for (int n = 0; n < NBEST; ++n) {
        float cv = tr + lpart[i][n];
        if (cv <= vv[NBEST - 1]) break;
        int kp = NBEST - 1;
        while (kp > 0 && vv[kp - 1] < cv) {
          vv[kp] = vv[kp - 1]; ii[kp] = ii[kp - 1]; --kp;
        }
        vv[kp] = cv; ii[kp] = i * NBEST + n;
      }
    }
    int p0a[NBEST];
#pragma unroll
    for (int n = 0; n < NBEST; ++n) p0a[n] = ii[n];
    out[(size_t)b * SS + SS - 1] = (float)(p0a[0] >> 3);
    int p = p0a[0];
    for (int t = SS - 2; t >= 0; --t) {
      int np;
      if (t == last_pos) np = p0a[p & 7];
      else np = (int)bps[((size_t)t * BB + b) * (TAG * NBEST) + p];
      out[(size_t)b * SS + t] = (float)(np >> 3);
      int m = mask[b * SS + t];
      p = np + p * (1 - m);
    }
  }
}

extern "C" void kernel_launch(void* const* d_in, const int* in_sizes, int n_in,
                              void* d_out, int out_size, void* d_ws, size_t ws_size,
                              hipStream_t stream) {
  (void)in_sizes; (void)n_in; (void)out_size; (void)ws_size;
  const float* x     = (const float*)d_in[0];
  const int*   mask  = (const int*)d_in[1];
  const float* wq    = (const float*)d_in[2];
  const float* wk    = (const float*)d_in[3];
  const float* wvv   = (const float*)d_in[4];
  const float* pw    = (const float*)d_in[5];
  const float* pb    = (const float*)d_in[6];
  const float* lng   = (const float*)d_in[7];
  const float* lnb   = (const float*)d_in[8];
  const float* w1    = (const float*)d_in[9];
  const float* b1    = (const float*)d_in[10];
  const float* w2    = (const float*)d_in[11];
  const float* b2    = (const float*)d_in[12];
  const float* trans = (const float*)d_in[13];

  char* ws = (char*)d_ws;
  _Float16* q    = (_Float16*)(ws + 0);
  _Float16* k    = (_Float16*)(ws + 12582912);
  _Float16* vT   = (_Float16*)(ws + 25165824);
  _Float16* attn = (_Float16*)(ws + 37748736);
  float*    post = (float*)   (ws + 50331648);   // f32, 100.7 MB
  _Float16* ln16 = (_Float16*)(ws + 0);          // aliases q/k/vT/attn (dead by then)
  _Float16* h1   = (_Float16*)(ws + 50331648);   // aliases post (dead by then)
  float*    lgts = (float*)   (ws + 75497472);
  short*    bpsw = (short*)   (ws + 79822848);
  float*    out  = (float*)d_out;

  k_qkv   <<<dim3(BB * HH, 4), 256, 0, stream>>>(x, wq, wk, wvv, q, k, vT);
  k_attn  <<<dim3(BB * HH, 4), 256, 0, stream>>>(q, k, vT, attn);
  k_proj  <<<dim3(256, 12),    256, 0, stream>>>(attn, pw, pb, x, post);
  k_ln    <<<dim3(4096),       256, 0, stream>>>(post, lng, lnb, ln16);
  k_lin1  <<<dim3(256, 6),     256, 0, stream>>>(ln16, w1, b1, h1);
  k_lin2  <<<dim3(256),        256, 0, stream>>>(h1, w2, b2, lgts);
  k_viterbi<<<dim3(BB),         64, 0, stream>>>(lgts, mask, trans, bpsw, out);
}
